// DonutSwinSelfAttention_17497696764139
// MI455X (gfx1250) — matrix operand order repaired and checked by reading, not compiled, not run on hardware
//
#include <hip/hip_runtime.h>
#include <hip/hip_bf16.h>

#define NHEADS 4
#define HDIM 32
#define DIM 128
#define SEQ 49
#define SEQP 64
#define MASKW 64

typedef _Float16 v16h   __attribute__((ext_vector_type(16)));
typedef _Float16 half8  __attribute__((ext_vector_type(8)));
typedef _Float16 half2v __attribute__((ext_vector_type(2)));
typedef float    v8f    __attribute__((ext_vector_type(8)));

// ---------------------------------------------------------------------------
// Prep: fp32 weights -> f16, packed [3][128][128] into workspace (L2 resident)
// ---------------------------------------------------------------------------
__global__ void convert_weights_kernel(const float* __restrict__ wq,
                                       const float* __restrict__ wk,
                                       const float* __restrict__ wv,
                                       _Float16* __restrict__ o) {
  int i = blockIdx.x * blockDim.x + threadIdx.x;
  if (i < DIM * DIM) {
    o[i]                 = (_Float16)wq[i];
    o[DIM * DIM + i]     = (_Float16)wk[i];
    o[2 * DIM * DIM + i] = (_Float16)wv[i];
  }
}

// A fragment (16x32 f16): lane half selects K groups {base..base+7, base+16..base+23}
__device__ __forceinline__ v16h frag_row_a(const _Float16* p0, int k0, int half_) {
  const _Float16* p = p0 + k0 + half_ * 8;
  half8 lo = *(const half8*)(p);
  half8 hi = *(const half8*)(p + 16);
  v16h r;
#pragma unroll
  for (int j = 0; j < 8; ++j) { r[j] = lo[j]; r[j + 8] = hi[j]; }
  return r;
}

// B fragment (32x16 f16) loaded from rows of B^T: lane half selects K groups {base..base+7, base+16..base+23}
__device__ __forceinline__ v16h frag_row_b(const _Float16* p0, int k0, int half_) {
  const _Float16* p = p0 + k0 + half_ * 8;
  half8 lo = *(const half8*)(p);
  half8 hi = *(const half8*)(p + 16);
  v16h r;
#pragma unroll
  for (int j = 0; j < 8; ++j) { r[j] = lo[j]; r[j + 8] = hi[j]; }
  return r;
}

// ---------------------------------------------------------------------------
// Main: one workgroup (8 wave32) per window
// ---------------------------------------------------------------------------
__launch_bounds__(256)
__global__ void swin_attention_kernel(const float* __restrict__ x,
                                      const float* __restrict__ amask,
                                      const _Float16* __restrict__ w3,
                                      const float* __restrict__ bq,
                                      const float* __restrict__ bk,
                                      const float* __restrict__ bv,
                                      const float* __restrict__ btab,
                                      const int* __restrict__ pidx,
                                      float* __restrict__ out) {
  // 128 KB LDS total; X|Q region is reused for probs P after scores are done.
  __shared__ __align__(16) _Float16 uXQP[2 * SEQP * DIM];          // 32 KB
  __shared__ __align__(16) _Float16 sK[SEQP * DIM];                // 16 KB
  __shared__ __align__(16) _Float16 sVT[DIM * SEQP];               // 16 KB (V transposed)
  __shared__ __align__(16) float    sS[NHEADS * SEQP * SEQP];      // 64 KB

  _Float16* sX = uXQP;
  _Float16* sQ = uXQP + SEQP * DIM;
  _Float16* sP = uXQP;  // [4][64][64], valid after __syncthreads() post-scores

  const int tid   = threadIdx.x;
  const int wv_   = (tid >> 5) & 7;
  const int lane  = tid & 31;
  const int ln16  = lane & 15;
  const int half_ = lane >> 4;

  // ---- Phase 1: stage X as f16, zero-pad rows 49..63 -----------------------
  const float* xw = x + (size_t)blockIdx.x * (SEQ * DIM);
  for (int i = tid; i < SEQP * DIM; i += 256) {
    int r = i >> 7, c = i & 127;
    sX[i] = (r < SEQ) ? (_Float16)xw[r * DIM + c] : (_Float16)0.0f;
  }
  __syncthreads();

  // ---- Phase 2: Q/K/V = X @ W^T + b  (96 tiles, 4 k-steps each) ------------
  // Q,K stored row-major; V stored transposed so PV B-fragments are contiguous.
#pragma unroll
  for (int mat = 0; mat < 3; ++mat) {
    const float* bp = (mat == 0) ? bq : ((mat == 1) ? bk : bv);
    const _Float16* wbase = w3 + mat * (DIM * DIM);
#pragma unroll
    for (int it = 0; it < 4; ++it) {
      int t  = wv_ + (it << 3);  // 0..31
      int mt = t & 3;
      int nt = t >> 2;           // 0..7
      const _Float16* arow = sX + (mt * 16 + ln16) * DIM;
      const _Float16* wrow = wbase + (nt * 16 + ln16) * DIM;  // row n of W = col n of W^T
      v8f acc = {};
#pragma unroll
      for (int kc = 0; kc < 4; ++kc) {
        v16h a = frag_row_a(arow, kc * 32, half_);
        v16h b = frag_row_b(wrow, kc * 32, half_);
        acc = __builtin_amdgcn_wmma_f32_16x16x32_f16(false, a, false, b,
                                                     (short)0, acc, false, false);
      }
      float bb = bp[nt * 16 + ln16];
      if (mat == 2) {
        // V^T[col][row]: the 8 D rows per lane are contiguous -> one b128 store
        half8 pk;
#pragma unroll
        for (int r = 0; r < 8; ++r) pk[r] = (_Float16)(acc[r] + bb);
        *(half8*)(sVT + (nt * 16 + ln16) * SEQP + mt * 16 + half_ * 8) = pk;
      } else {
        _Float16* dst = (mat == 0) ? sQ : sK;
#pragma unroll
        for (int r = 0; r < 8; ++r) {
          int row = mt * 16 + r + half_ * 8;
          dst[row * DIM + nt * 16 + ln16] = (_Float16)(acc[r] + bb);
        }
      }
    }
  }
  __syncthreads();

  // ---- Phase 3: scores = Q K^T per head (K = HEAD_DIM = 32 -> 1 WMMA/tile) -
#pragma unroll
  for (int it = 0; it < 8; ++it) {
    int t  = wv_ + (it << 3);   // 0..63
    int h  = (t >> 4) & 3;
    int mt = (t >> 2) & 3;
    int nt = t & 3;
    const _Float16* qrow = sQ + (mt * 16 + ln16) * DIM + h * HDIM;
    const _Float16* krow = sK + (nt * 16 + ln16) * DIM + h * HDIM;  // rows of K = cols of K^T
    v16h a = frag_row_a(qrow, 0, half_);
    v16h b = frag_row_b(krow, 0, half_);
    v8f acc = {};
    acc = __builtin_amdgcn_wmma_f32_16x16x32_f16(false, a, false, b,
                                                 (short)0, acc, false, false);
#pragma unroll
    for (int r = 0; r < 8; ++r) {
      int row = mt * 16 + r + half_ * 8;
      sS[(h * SEQP + row) * SEQP + nt * 16 + ln16] = acc[r];
    }
  }
  __syncthreads();

  // ---- Phase 4: softmax rows (scale + rel-bias gather + image mask) --------
  {
    int h = tid >> 6;   // 0..3
    int i = tid & 63;   // 0..63
    _Float16* prow = sP + (h * SEQP + i) * SEQP;
    if (i < SEQ) {
      const float scale = 0.17677669529663687f;  // 1/sqrt(32)
      float* srow = sS + (h * SEQP + i) * SEQP;
      const float* mrow = amask + (size_t)(blockIdx.x % MASKW) * (SEQ * SEQ) + i * SEQ;
      const int* pr = pidx + i * SEQ;
      float mx = -3.0e38f;
      for (int j = 0; j < SEQ; ++j) {
        float v = srow[j] * scale + btab[pr[j] * NHEADS + h] + mrow[j];
        mx = fmaxf(mx, v);
      }
      float sum = 0.0f;
      for (int j = 0; j < SEQ; ++j) {
        float v = srow[j] * scale + btab[pr[j] * NHEADS + h] + mrow[j];
        float e = __expf(v - mx);
        sum += e;
        srow[j] = e;  // stash exp for normalization pass
      }
      float inv = 1.0f / sum;
#pragma unroll
      for (int jj = 0; jj < SEQP; jj += 2) {  // paired b32 stores, pads cols 49..63
        half2v p;
        p[0] = (jj     < SEQ) ? (_Float16)(srow[jj] * inv)     : (_Float16)0.0f;
        p[1] = (jj + 1 < SEQ) ? (_Float16)(srow[jj + 1] * inv) : (_Float16)0.0f;
        *(half2v*)(prow + jj) = p;
      }
    } else {
      half8 z = {};
#pragma unroll
      for (int jj = 0; jj < SEQP; jj += 8) *(half8*)(prow + jj) = z;
    }
  }
  __syncthreads();

  // ---- Phase 5: ctx = P @ V  (K padded to 64 -> 2 WMMAs per tile) ----------
  const size_t obase = (size_t)blockIdx.x * (SEQ * DIM);
#pragma unroll
  for (int it = 0; it < 4; ++it) {
    int t  = wv_ + (it << 3);   // 0..31
    int h  = (t >> 3) & 3;
    int mt = (t >> 1) & 3;
    int nt = t & 1;
    const _Float16* arow  = sP + (h * SEQP + mt * 16 + ln16) * SEQP;
    int ncol = h * HDIM + nt * 16 + ln16;
    const _Float16* vtrow = sVT + ncol * SEQP;  // row of V^T = column of V
    v8f acc = {};
#pragma unroll
    for (int kc = 0; kc < 2; ++kc) {
      v16h a = frag_row_a(arow, kc * 32, half_);
      v16h b = frag_row_b(vtrow, kc * 32, half_);
      acc = __builtin_amdgcn_wmma_f32_16x16x32_f16(false, a, false, b,
                                                   (short)0, acc, false, false);
    }
#pragma unroll
    for (int r = 0; r < 8; ++r) {
      int row = mt * 16 + r + half_ * 8;
      if (row < SEQ) out[obase + (size_t)row * DIM + ncol] = acc[r];
    }
  }
}

// ---------------------------------------------------------------------------
extern "C" void kernel_launch(void* const* d_in, const int* in_sizes, int n_in,
                              void* d_out, int out_size, void* d_ws, size_t ws_size,
                              hipStream_t stream) {
  (void)n_in; (void)out_size; (void)ws_size;
  const float* x  = (const float*)d_in[0];
  const float* am = (const float*)d_in[1];
  const float* wq = (const float*)d_in[2];
  const float* bq = (const float*)d_in[3];
  const float* wk = (const float*)d_in[4];
  const float* bk = (const float*)d_in[5];
  const float* wv = (const float*)d_in[6];
  const float* bv = (const float*)d_in[7];
  const float* bt = (const float*)d_in[8];
  const int*   pi = (const int*)d_in[9];
  float* out = (float*)d_out;
  _Float16* w3 = (_Float16*)d_ws;  // 3*128*128 f16 = 96 KB

  convert_weights_kernel<<<(DIM * DIM + 255) / 256, 256, 0, stream>>>(wq, wk, wv, w3);

  int nblocks = in_sizes[0] / (SEQ * DIM);  // 8192 windows
  swin_attention_kernel<<<nblocks, 256, 0, stream>>>(x, am, w3, bq, bk, bv, bt, pi, out);
}
